// LLMAttention_6279242186938
// MI455X (gfx1250) — matrix operand order repaired and checked by reading, not compiled, not run on hardware
//
#include <hip/hip_runtime.h>
#include <hip/hip_bf16.h>
#include <math.h>

typedef _Float16 half_t;
typedef __attribute__((ext_vector_type(16))) _Float16 v16h;
typedef __attribute__((ext_vector_type(8)))  _Float16 v8h;
typedef __attribute__((ext_vector_type(8)))  float    v8f;

#define B_    2
#define H_    16
#define L_    4096
#define D_    1024
#define DH_   64
#define N3_   3072
#define BHL_  (B_*H_*L_)     // 131072
#define MROWS_ (B_*L_)       // 8192
#define NBLK_ 16
#define SAMP_ 256

static __device__ inline v8f vzero8f() {
  v8f z = {0.f,0.f,0.f,0.f,0.f,0.f,0.f,0.f};
  return z;
}

// A-matrix 16x32 f16 fragment:
// lane M = lane%16, h = lane/16; elems 0..15 -> K = h*16 + i.
// 'row' points at A[M][0] (row-major, >=32 halfs).
static __device__ inline v16h frag_a16(const half_t* row, int h) {
  return *(const v16h*)(row + h * 16);
}

// Async global->LDS copy of 32 bytes (two B128 ops; the immediate offset is
// applied to BOTH the LDS and the global address per the CDNA5 ISA).
static __device__ inline void async_copy32(unsigned lds_addr, const half_t* gptr) {
  asm volatile(
      "global_load_async_to_lds_b128 %0, %1, off\n\t"
      "global_load_async_to_lds_b128 %0, %1, off offset:16"
      :
      : "v"(lds_addr), "v"(gptr)
      : "memory");
}

static __device__ inline void wait_async_le4() {
  asm volatile("s_wait_asynccnt 0x4" ::: "memory");
}
static __device__ inline void wait_async_0() {
  asm volatile("s_wait_asynccnt 0x0" ::: "memory");
}

// ---------------------------------------------------------------------------
// f32 -> f16 conversion
// ---------------------------------------------------------------------------
__global__ void cvt_f32_f16(const float* __restrict__ in, half_t* __restrict__ out, int n) {
  int i = blockIdx.x * blockDim.x + threadIdx.x;
  if (i < n) out[i] = (half_t)in[i];
}

// ---------------------------------------------------------------------------
// WMMA GEMM: C[m][n] = sum_k A[m][k] * W[n][k] + bias[n]
// Double-buffered LDS tiles filled by async global->LDS copies.
// MODE 0: scatter into q/k/v f16 buffers laid out (B,H,L,64)
// MODE 1: write f32 to outF (row-major MxN)
// ---------------------------------------------------------------------------
template <int MODE>
__global__ __launch_bounds__(256) void gemm_wmma(
    const half_t* __restrict__ A, const half_t* __restrict__ W,
    const float* __restrict__ bias, int M, int N, int K,
    float* __restrict__ outF, half_t* __restrict__ qd,
    half_t* __restrict__ kd, half_t* __restrict__ vd) {
  __shared__ half_t As[2][128][32];
  __shared__ half_t Ws[2][128][32];

  const int wave = threadIdx.x >> 5;
  const int lane = threadIdx.x & 31;
  const int Nl = lane & 15;
  const int hh = lane >> 4;
  const int wm = wave >> 1;      // 0..3  (m direction)
  const int wn = wave & 1;       // 0..1  (n direction)
  const int m0 = blockIdx.y * 128;
  const int n0 = blockIdx.x * 128;

  v8f acc[2][4];
#pragma unroll
  for (int i = 0; i < 2; ++i)
#pragma unroll
    for (int j = 0; j < 4; ++j) acc[i][j] = vzero8f();

  const int lrow = threadIdx.x >> 1;   // 0..127
  const int seg = threadIdx.x & 1;     // 0..1
  const int ntiles = K >> 5;

  const half_t* gA = &A[(size_t)(m0 + lrow) * K + seg * 16];
  const half_t* gW = &W[(size_t)(n0 + lrow) * K + seg * 16];
  unsigned ldsA[2], ldsW[2];
  ldsA[0] = (unsigned)(uintptr_t)&As[0][lrow][seg * 16];
  ldsA[1] = (unsigned)(uintptr_t)&As[1][lrow][seg * 16];
  ldsW[0] = (unsigned)(uintptr_t)&Ws[0][lrow][seg * 16];
  ldsW[1] = (unsigned)(uintptr_t)&Ws[1][lrow][seg * 16];

  // prologue: fill buffer 0 with tile 0
  async_copy32(ldsA[0], gA);
  async_copy32(ldsW[0], gW);

  for (int kt = 0; kt < ntiles; ++kt) {
    const int cur = kt & 1;
    __syncthreads();  // everyone done reading buf[cur^1] -> safe to overwrite
    if (kt + 1 < ntiles) {
      async_copy32(ldsA[cur ^ 1], gA + (size_t)(kt + 1) * 32);
      async_copy32(ldsW[cur ^ 1], gW + (size_t)(kt + 1) * 32);
      wait_async_le4();  // older 4 (current tile) complete; new 4 in flight
    } else {
      wait_async_0();
    }
    __syncthreads();  // buf[cur] visible to all waves

    v16h af[2];
#pragma unroll
    for (int i = 0; i < 2; ++i)
      af[i] = frag_a16(&As[cur][wm * 32 + i * 16 + Nl][0], hh);
    v16h bf[4];
#pragma unroll
    for (int j = 0; j < 4; ++j)
      bf[j] = *(const v16h*)&Ws[cur][wn * 64 + j * 16 + Nl][hh * 16];

#pragma unroll
    for (int i = 0; i < 2; ++i)
#pragma unroll
      for (int j = 0; j < 4; ++j)
        acc[i][j] = __builtin_amdgcn_wmma_f32_16x16x32_f16(
            false, af[i], false, bf[j], (short)0, acc[i][j], false, false);
  }

  // Epilogue. C layout: VGPR r holds row M = r + 8*h, col N = lane%16.
#pragma unroll
  for (int i = 0; i < 2; ++i) {
#pragma unroll
    for (int j = 0; j < 4; ++j) {
#pragma unroll
      for (int r = 0; r < 8; ++r) {
        int gm = m0 + wm * 32 + i * 16 + r + 8 * hh;
        int gn = n0 + wn * 64 + j * 16 + Nl;
        float val = acc[i][j][r] + bias[gn];
        if (MODE == 0) {
          int t = gn >> 10;            // 0=q 1=k 2=v
          int hd = gn & 1023;
          int hidx = hd >> 6, c = hd & 63;
          int b = gm >> 12, l = gm & 4095;
          half_t* dst = (t == 0) ? qd : (t == 1) ? kd : vd;
          dst[((size_t)((b << 4) + hidx) * L_ + l) * DH_ + c] = (half_t)val;
        } else {
          outF[(size_t)gm * N + gn] = val;
        }
      }
    }
  }
}

// ---------------------------------------------------------------------------
// LSH hash: bucket = Gray(bits of sign(q . proj_dir))
// ---------------------------------------------------------------------------
__global__ void lsh_hash_kernel(const half_t* __restrict__ q,
                                const half_t* __restrict__ k,
                                const float* __restrict__ proj,
                                int* __restrict__ qh, int* __restrict__ kh,
                                int total) {
  int i = blockIdx.x * blockDim.x + threadIdx.x;
  if (i >= total) return;
  const half_t* qr = q + (size_t)i * DH_;
  const half_t* kr = k + (size_t)i * DH_;
  float dq[7], dk[7];
#pragma unroll
  for (int p = 0; p < 7; ++p) { dq[p] = 0.f; dk[p] = 0.f; }
  for (int d = 0; d < DH_; ++d) {
    float qv = (float)qr[d];
    float kv = (float)kr[d];
#pragma unroll
    for (int p = 0; p < 7; ++p) {
      float w = proj[d * 7 + p];
      dq[p] += qv * w;
      dk[p] += kv * w;
    }
  }
  int bq = 0, bk = 0;
#pragma unroll
  for (int p = 0; p < 7; ++p) {
    bq |= (dq[p] > 0.f) << p;
    bk |= (dk[p] > 0.f) << p;
  }
  qh[i] = bq ^ (bq >> 1);   // binary-reflected Gray code == reference _PERM
  kh[i] = bk ^ (bk >> 1);
}

// ---------------------------------------------------------------------------
// Stable counting sort (argsort) of 4096 7-bit keys per (b,h). 1 WG per (b,h).
// ---------------------------------------------------------------------------
__global__ __launch_bounds__(64) void csort_kernel(const int* __restrict__ hash,
                                                   int* __restrict__ sidx) {
  __shared__ int cnt[64][128];
  __shared__ int tot[128];
  __shared__ int off[128];
  const int t = threadIdx.x;
  const int* h = hash + (size_t)blockIdx.x * L_;
  int* out = sidx + (size_t)blockIdx.x * L_;

  for (int b = 0; b < 128; ++b) cnt[t][b] = 0;
  __syncthreads();
  const int base = t * 64;
  for (int e = 0; e < 64; ++e) cnt[t][h[base + e]]++;
  __syncthreads();
  for (int bb = t; bb < 128; bb += 64) {
    int s = 0;
    for (int r = 0; r < 64; ++r) s += cnt[r][bb];
    tot[bb] = s;
  }
  __syncthreads();
  if (t == 0) {
    int acc = 0;
    for (int b = 0; b < 128; ++b) { off[b] = acc; acc += tot[b]; }
  }
  __syncthreads();
  for (int bb = t; bb < 128; bb += 64) {
    int run = off[bb];
    for (int r = 0; r < 64; ++r) {
      int c = cnt[r][bb];
      cnt[r][bb] = run;
      run += c;
    }
  }
  __syncthreads();
  for (int e = 0; e < 64; ++e) {
    int idx = base + e;
    int key = h[idx];
    int pos = cnt[t][key]++;
    out[pos] = idx;
  }
}

// ---------------------------------------------------------------------------
// Gather sorted q/k/v rows
// ---------------------------------------------------------------------------
__global__ void gather_kernel(const half_t* __restrict__ q,
                              const half_t* __restrict__ k,
                              const half_t* __restrict__ v,
                              const int* __restrict__ qs,
                              const int* __restrict__ ks,
                              half_t* __restrict__ q_s,
                              half_t* __restrict__ k_s,
                              half_t* __restrict__ v_s) {
  int i = blockIdx.x * blockDim.x + threadIdx.x;   // 0..BHL_-1 (sorted pos)
  int bh = i >> 12;
  size_t dst = (size_t)i * DH_;
  size_t qsrc = ((size_t)(bh << 12) + qs[i]) * DH_;
  size_t ksrc = ((size_t)(bh << 12) + ks[i]) * DH_;
#pragma unroll
  for (int c = 0; c < 4; ++c) {
    *(v16h*)&q_s[dst + c * 16] = *(const v16h*)&q[qsrc + c * 16];
    *(v16h*)&k_s[dst + c * 16] = *(const v16h*)&k[ksrc + c * 16];
    *(v16h*)&v_s[dst + c * 16] = *(const v16h*)&v[ksrc + c * 16];
  }
}

// ---------------------------------------------------------------------------
// Attention over 256 keys per query block, flash-style single pass.
// RESIDUAL=0: block-diagonal (keys = own block), write attn_b + lse_b
// RESIDUAL=1: sampled keys + block mask; fused combine + un-sort scatter
// grid = B*H*NBLK_, 256 threads (8 waves, 32 query rows each)
// ---------------------------------------------------------------------------
template <int RESIDUAL>
__global__ __launch_bounds__(256) void attn_wmma(
    const half_t* __restrict__ q_s, const half_t* __restrict__ k_s,
    const half_t* __restrict__ v_s, const int* __restrict__ sampled,
    const int* __restrict__ qsidx, float* __restrict__ attn_b,
    float* __restrict__ lse_b, half_t* __restrict__ attn_x) {
  __shared__ half_t Ks[128][64];        // 16 KB   [key][d]
  __shared__ half_t Vt[64][128];        // 16 KB   [d][key]
  __shared__ half_t Pb[8][32][32];      // 16 KB   per-wave P chunk
  __shared__ float biasL[256];

  const int bh = blockIdx.x >> 4;
  const int blk = blockIdx.x & 15;
  const int wave = threadIdx.x >> 5;
  const int lane = threadIdx.x & 31;
  const int Nl = lane & 15;
  const int hh = lane >> 4;
  const int qbase = (bh << 12) + (blk << 8);
  const int rowbase = wave * 32;

  if (RESIDUAL) {
    int sidx = sampled[(bh << 8) + (int)threadIdx.x];
    biasL[threadIdx.x] = ((sidx >> 8) == blk) ? -1e30f : 0.0f;
  }

  // preload Q fragments: 2 m-tiles x 2 d-chunks
  v16h qf[2][2];
#pragma unroll
  for (int i = 0; i < 2; ++i) {
    const half_t* row = q_s + (size_t)(qbase + rowbase + i * 16 + Nl) * DH_;
#pragma unroll
    for (int dc = 0; dc < 2; ++dc) qf[i][dc] = frag_a16(row + dc * 32, hh);
  }

  v8f o[2][4];
  float mrow[2][8], srow[2][8];
#pragma unroll
  for (int i = 0; i < 2; ++i) {
#pragma unroll
    for (int j = 0; j < 4; ++j) o[i][j] = vzero8f();
#pragma unroll
    for (int r = 0; r < 8; ++r) { mrow[i][r] = -1e30f; srow[i][r] = 0.f; }
  }

  const int r2 = threadIdx.x >> 1, seg = threadIdx.x & 1;
  const unsigned ldsK = (unsigned)(uintptr_t)&Ks[r2][seg * 32];

  for (int kc = 0; kc < 2; ++kc) {
    __syncthreads();
    {  // stage 128 keys: K via async global->LDS, V transposed via VGPRs
      int keyl = kc * 128 + r2;
      int src = RESIDUAL ? ((bh << 12) + sampled[(bh << 8) + keyl])
                         : (qbase + keyl);
      async_copy32(ldsK, k_s + (size_t)src * DH_ + seg * 32);
      const half_t* vp = v_s + (size_t)src * DH_ + seg * 32;
      half_t tmpv[32] __attribute__((aligned(32)));
      *(v16h*)&tmpv[0] = *(const v16h*)vp;
      *(v16h*)&tmpv[16] = *(const v16h*)(vp + 16);
      for (int d = 0; d < 32; ++d) Vt[seg * 32 + d][r2] = tmpv[d];
      wait_async_0();
    }
    __syncthreads();

    for (int cc = 0; cc < 4; ++cc) {
      const int key0 = cc * 32;
      // --- S = Q K^T (2 m-tiles x 2 n-tiles, K-dim = 64 in 2 chunks) ---
      v8f sac[2][2];
#pragma unroll
      for (int i = 0; i < 2; ++i)
#pragma unroll
        for (int jn = 0; jn < 2; ++jn) sac[i][jn] = vzero8f();
#pragma unroll
      for (int dc = 0; dc < 2; ++dc) {
#pragma unroll
        for (int jn = 0; jn < 2; ++jn) {
          v16h bf = *(const v16h*)&Ks[key0 + jn * 16 + Nl][dc * 32 + hh * 16];
#pragma unroll
          for (int i = 0; i < 2; ++i)
            sac[i][jn] = __builtin_amdgcn_wmma_f32_16x16x32_f16(
                false, qf[i][dc], false, bf, (short)0, sac[i][jn], false, false);
        }
      }

      // --- online softmax stats + P (f16) into per-wave LDS ---
#pragma unroll
      for (int i = 0; i < 2; ++i) {
        float bias0 = 0.f, bias1 = 0.f;
        if (RESIDUAL) {
          bias0 = biasL[kc * 128 + key0 + Nl];
          bias1 = biasL[kc * 128 + key0 + 16 + Nl];
        }
#pragma unroll
        for (int r = 0; r < 8; ++r) {
          float sv0 = sac[i][0][r] * 0.125f + bias0;
          float sv1 = sac[i][1][r] * 0.125f + bias1;
          float cm = fmaxf(sv0, sv1);
#pragma unroll
          for (int x = 1; x < 16; x <<= 1)
            cm = fmaxf(cm, __shfl_xor(cm, x, 32));
          float mn = fmaxf(mrow[i][r], cm);
          float fs = expf(mrow[i][r] - mn);
          float e0 = expf(sv0 - mn);
          float e1 = expf(sv1 - mn);
          float ps = e0 + e1;
#pragma unroll
          for (int x = 1; x < 16; x <<= 1) ps += __shfl_xor(ps, x, 32);
          srow[i][r] = srow[i][r] * fs + ps;
          mrow[i][r] = mn;
#pragma unroll
          for (int j = 0; j < 4; ++j) o[i][j][r] *= fs;
          int mloc = i * 16 + r + 8 * hh;
          Pb[wave][mloc][Nl] = (half_t)e0;
          Pb[wave][mloc][16 + Nl] = (half_t)e1;
        }
      }
      __syncthreads();

      // --- O += P V (K-dim = 32 keys of this chunk) ---
#pragma unroll
      for (int i = 0; i < 2; ++i) {
        v16h pa = frag_a16(&Pb[wave][i * 16 + Nl][0], hh);
#pragma unroll
        for (int j = 0; j < 4; ++j) {
          v16h bv = *(const v16h*)&Vt[j * 16 + Nl][key0 + hh * 16];
          o[i][j] = __builtin_amdgcn_wmma_f32_16x16x32_f16(
              false, pa, false, bv, (short)0, o[i][j], false, false);
        }
      }
      __syncthreads();
    }
  }

  // --- epilogue ---
#pragma unroll
  for (int i = 0; i < 2; ++i) {
#pragma unroll
    for (int r = 0; r < 8; ++r) {
      int row = qbase + rowbase + i * 16 + r + 8 * hh;
      float inv = 1.0f / srow[i][r];
      float lse = mrow[i][r] + logf(srow[i][r]);
      if (!RESIDUAL) {
#pragma unroll
        for (int j = 0; j < 4; ++j)
          attn_b[(size_t)row * DH_ + j * 16 + Nl] = o[i][j][r] * inv;
        if (Nl == 0) lse_b[row] = lse;
      } else {
        float lb = lse_b[row];
        float lr = lse + 2.7725887222397811f;  // log(N / SAMPLE) = log 16
        float c = 1.f / (1.f + expf(-(lb - lr)));
        int l_orig = qsidx[row];
        int b = bh >> 4, hd = bh & 15;
        size_t dst = ((size_t)((b << 12) + l_orig)) * (size_t)D_ + hd * DH_;
#pragma unroll
        for (int j = 0; j < 4; ++j) {
          int col = j * 16 + Nl;
          float ab = attn_b[(size_t)row * DH_ + col];
          float comb = c * ab + (1.f - c) * (o[i][j][r] * inv);
          attn_x[dst + col] = (half_t)comb;
        }
      }
    }
  }
}

// ---------------------------------------------------------------------------
extern "C" void kernel_launch(void* const* d_in, const int* in_sizes, int n_in,
                              void* d_out, int out_size, void* d_ws,
                              size_t ws_size, hipStream_t stream) {
  (void)in_sizes; (void)n_in; (void)out_size; (void)ws_size;
  const float* x = (const float*)d_in[0];
  const float* Wqkv = (const float*)d_in[1];
  const float* bqkv = (const float*)d_in[2];
  const float* Wproj = (const float*)d_in[3];
  const float* bproj = (const float*)d_in[4];
  const float* projd = (const float*)d_in[5];
  const int* sampled = (const int*)d_in[6];

  char* p = (char*)d_ws;
  auto take = [&](size_t bytes) {
    char* r = p;
    p += (bytes + 255) & ~(size_t)255;
    return r;
  };

  half_t* xh = (half_t*)take((size_t)MROWS_ * D_ * 2);
  half_t* wqkvh = (half_t*)take((size_t)N3_ * D_ * 2);
  half_t* wprojh = (half_t*)take((size_t)D_ * D_ * 2);
  half_t* q = (half_t*)take((size_t)BHL_ * DH_ * 2);
  half_t* k = (half_t*)take((size_t)BHL_ * DH_ * 2);
  half_t* v = (half_t*)take((size_t)BHL_ * DH_ * 2);
  int* qh = (int*)take((size_t)BHL_ * 4);
  int* kh = (int*)take((size_t)BHL_ * 4);
  int* qsidx = (int*)take((size_t)BHL_ * 4);
  int* ksidx = (int*)take((size_t)BHL_ * 4);
  half_t* q_s = (half_t*)take((size_t)BHL_ * DH_ * 2);
  half_t* k_s = (half_t*)take((size_t)BHL_ * DH_ * 2);
  half_t* v_s = (half_t*)take((size_t)BHL_ * DH_ * 2);
  float* attnb = (float*)take((size_t)BHL_ * DH_ * 4);
  float* lseb = (float*)take((size_t)BHL_ * 4);
  half_t* attnx = (half_t*)take((size_t)MROWS_ * D_ * 2);

  // 1) f32 -> f16 conversions
  cvt_f32_f16<<<(MROWS_ * D_ + 255) / 256, 256, 0, stream>>>(x, xh, MROWS_ * D_);
  cvt_f32_f16<<<(N3_ * D_ + 255) / 256, 256, 0, stream>>>(Wqkv, wqkvh, N3_ * D_);
  cvt_f32_f16<<<(D_ * D_ + 255) / 256, 256, 0, stream>>>(Wproj, wprojh, D_ * D_);

  // 2) QKV projection (WMMA), scattered into q/k/v (B,H,L,64) f16
  gemm_wmma<0><<<dim3(N3_ / 128, MROWS_ / 128), 256, 0, stream>>>(
      xh, wqkvh, bqkv, MROWS_, N3_, D_, nullptr, q, k, v);

  // 3) LSH hashes
  lsh_hash_kernel<<<BHL_ / 256, 256, 0, stream>>>(q, k, projd, qh, kh, BHL_);

  // 4) stable counting argsort per (b,h)
  csort_kernel<<<B_ * H_, 64, 0, stream>>>(qh, qsidx);
  csort_kernel<<<B_ * H_, 64, 0, stream>>>(kh, ksidx);

  // 5) gather sorted q/k/v
  gather_kernel<<<BHL_ / 256, 256, 0, stream>>>(q, k, v, qsidx, ksidx, q_s, k_s, v_s);

  // 6) block-diagonal attention (WMMA)
  attn_wmma<0><<<B_ * H_ * NBLK_, 256, 0, stream>>>(q_s, k_s, v_s, sampled,
                                                    qsidx, attnb, lseb, attnx);

  // 7) sampled residual attention + combine + un-sort (WMMA)
  attn_wmma<1><<<B_ * H_ * NBLK_, 256, 0, stream>>>(q_s, k_s, v_s, sampled,
                                                    qsidx, attnb, lseb, attnx);

  // 8) output projection (WMMA) -> f32 d_out
  gemm_wmma<1><<<dim3(D_ / 128, MROWS_ / 128), 256, 0, stream>>>(
      attnx, wprojh, bproj, MROWS_, D_, D_, (float*)d_out, nullptr, nullptr,
      nullptr);
}
